// GRU_68341519614597
// MI455X (gfx1250) — matrix operand order repaired and checked by reading, not compiled, not run on hardware
//
#include <hip/hip_runtime.h>
#include <hip/hip_bf16.h>

// 2-layer GRU, H=256, seq=64, rows M=2048 (=16*128). Row-independent recurrence
// -> 128 blocks of one 16-row WMMA tile each, persistent over 64 time steps.
// Matmuls use v_wmma_f32_16x16x32_bf16 (fp32 accum). Weights pre-swizzled to
// bf16 WMMA B-fragment layout in d_ws once per launch and streamed from L2
// every step. All global traffic uses addrspace(1) pointers so the backend
// emits global_load/global_store (LOADcnt only) instead of flat ops (which
// would also tick DScnt and falsely serialize against the LDS A-frag loads).

#define HSZ   256
#define SEQ   64
#define MROWS 2048
#define G3H   768

#define AS1 __attribute__((address_space(1)))

typedef __attribute__((ext_vector_type(16))) __bf16 v16bf;
typedef __attribute__((ext_vector_type(8)))  float  v8f;

typedef const AS1 __bf16* cgbfp;   // global const bf16*
typedef const AS1 v16bf* cgv16p;   // global const v16bf*
typedef const AS1 float* cgfp;     // global const float*
typedef AS1 float*       gfp;      // global float*

// d_ws layout (bf16 elements): swizzled B-fragments, frag = 512 elems (32x16),
// ordered frag[n_tile][k_tile], lane-major inside.
#define WS_WI0 0         // K=64,  nKt=2 : 48*2*512   = 49152
#define WS_WH0 49152     // K=256, nKt=8 : 48*8*512   = 196608
#define WS_WI1 245760
#define WS_WH1 442368
#define WS_TOTAL 638976  // *2 bytes = 1.25 MB of d_ws

__global__ void gru_swizzle_weights(const float* __restrict__ wi0,
                                    const float* __restrict__ wh0,
                                    const float* __restrict__ wi1,
                                    const float* __restrict__ wh1,
                                    __bf16* __restrict__ ws) {
  int idx = blockIdx.x * 256 + threadIdx.x;
  if (idx >= WS_TOTAL) return;
  const float* src; int base, nKt;
  if (idx < WS_WH0)      { src = wi0; base = WS_WI0; nKt = 2; }
  else if (idx < WS_WI1) { src = wh0; base = WS_WH0; nKt = 8; }
  else if (idx < WS_WH1) { src = wi1; base = WS_WI1; nKt = 8; }
  else                   { src = wh1; base = WS_WH1; nKt = 8; }
  int j = idx - base;
  int k = j / G3H;          // contraction dim
  int n = j % G3H;          // gate-column dim
  // B-fragment (32x16, 16-bit): lane = ((K>>3)&1 ? 16:0) + N%16 ; elem = K%8 + (K%32>=16 ? 8:0)
  int lane = (((k >> 3) & 1) ? 16 : 0) + (n & 15);
  int e    = (k & 7) + ((k & 16) ? 8 : 0);
  int frag = (n >> 4) * nKt + (k >> 5);
  ws[base + frag * 512 + lane * 16 + e] = (__bf16)src[j];
}

__device__ __forceinline__ float sigm_(float v) {
  return 1.0f / (1.0f + __expf(-v));
}

__device__ __forceinline__ v8f wmma_bf16(v16bf a, v16bf b, v8f c) {
  return __builtin_amdgcn_wmma_f32_16x16x32_bf16(false, a, false, b, (short)0,
                                                 c, false, false);
}

// A (16 x 32*NKT, LDS A-fragments) @ W (gate-major column slices of this wave):
// gate 0 -> accR, gate 1 -> accZ, gate 2 -> accN. Callers pass the SAME
// accR/accZ for the input-gemm and hidden-gemm (r,z only need the sum), and
// DIFFERENT accN (n gate needs inn and hn separately).
template <int NKT>
__device__ __forceinline__ void gemm_gru(const __bf16* As, cgbfp W,
                                         int wave, int lane,
                                         v8f accR[2], v8f accZ[2], v8f accN[2]) {
#pragma unroll
  for (int kt = 0; kt < NKT; ++kt) {
    v16bf a = *(const v16bf*)(As + kt * 512 + lane * 16);
#pragma unroll
    for (int u = 0; u < 2; ++u) {
      int nt = wave * 2 + u;  // n-tile within one gate's 16 tiles
      v16bf b0 = *(cgv16p)(W + (((0  + nt) * NKT + kt) * 512) + lane * 16);
      accR[u] = wmma_bf16(a, b0, accR[u]);
      v16bf b1 = *(cgv16p)(W + (((16 + nt) * NKT + kt) * 512) + lane * 16);
      accZ[u] = wmma_bf16(a, b1, accZ[u]);
      v16bf b2 = *(cgv16p)(W + (((32 + nt) * NKT + kt) * 512) + lane * 16);
      accN[u] = wmma_bf16(a, b2, accN[u]);
    }
  }
}

__global__ __launch_bounds__(256)
void gru_persistent(const float* __restrict__ x,
                    const float* __restrict__ bi0, const float* __restrict__ bh0,
                    const float* __restrict__ bi1, const float* __restrict__ bh1,
                    const __bf16* __restrict__ ws,
                    float* __restrict__ out) {
  __shared__ __attribute__((aligned(32))) __bf16 hA0[8 * 512];  // h0 as A-frags
  __shared__ __attribute__((aligned(32))) __bf16 hA1[8 * 512];  // h1 as A-frags
  __shared__ __attribute__((aligned(32))) __bf16 xA[2 * 512];   // x_t as A-frags

  const int tid    = threadIdx.x;
  const int lane   = tid & 31;
  const int wave   = tid >> 5;             // 0..7: owns h-cols [32w,32w+32)
  const int col16  = lane & 15;            // C-frag: N within n-tile
  const int rowOff = (lane < 16) ? 0 : 8;  // C-frag: M base
  const int R0     = blockIdx.x * 16;      // global row base of this tile

  cgfp xg  = (cgfp)x;
  gfp  og  = (gfp)out;
  cgfp gbi0 = (cgfp)bi0, gbh0 = (cgfp)bh0, gbi1 = (cgfp)bi1, gbh1 = (cgfp)bh1;

  for (int i = tid; i < 8 * 512; i += 256) {
    hA0[i] = (__bf16)0.0f;
    hA1[i] = (__bf16)0.0f;
  }

  // Per-lane gate biases for this wave's column slice. r,z use bi+bh folded.
  float vbr0[2], vbz0[2], vbin0[2], vbhn0[2];
  float vbr1[2], vbz1[2], vbin1[2], vbhn1[2];
#pragma unroll
  for (int u = 0; u < 2; ++u) {
    int n = wave * 32 + u * 16 + col16;
    vbr0[u]  = gbi0[n] + gbh0[n];
    vbz0[u]  = gbi0[HSZ + n] + gbh0[HSZ + n];
    vbin0[u] = gbi0[2 * HSZ + n];
    vbhn0[u] = gbh0[2 * HSZ + n];
    vbr1[u]  = gbi1[n] + gbh1[n];
    vbz1[u]  = gbi1[HSZ + n] + gbh1[HSZ + n];
    vbin1[u] = gbi1[2 * HSZ + n];
    vbhn1[u] = gbh1[2 * HSZ + n];
  }

  v8f h0r[2] = {};  // fp32 state, C-fragment layout (this wave's 16x32 slice)
  v8f h1r[2] = {};
  __syncthreads();

  for (int t = 0; t < SEQ; ++t) {
    // Launder the weight base pointer so the compiler cannot hoist the
    // (loop-invariant) weight-fragment loads out of the time loop and spill.
    unsigned long long wsa = (unsigned long long)ws;
    asm volatile("" : "+s"(wsa));
    cgbfp wsl = (cgbfp)wsa;

    // ---- cooperative load of x_t tile (16x64 fp32 -> 2 bf16 A-fragments) ----
    {
      cgfp xs = xg + ((size_t)t * MROWS + R0) * 64;
#pragma unroll
      for (int i = 0; i < 4; ++i) {
        int idx = tid + 256 * i;             // 0..1023
        int m = idx >> 6, k = idx & 63;
        float v = xs[m * 64 + k];
        int kk = k & 31;
        int lp = m + (((kk >> 3) & 1) ? 16 : 0);
        int e  = (kk & 7) + ((kk >> 4) ? 8 : 0);
        xA[(k >> 5) * 512 + lp * 16 + e] = (__bf16)v;
      }
    }
    __syncthreads();  // xA ready; prev-iter hA1 writes visible

    // ---- layer 0: r/z accumulate gi+gh together; n keeps i/h split ----
    v8f aR[2] = {}, aZ[2] = {}, aNi[2] = {}, aNh[2] = {};
    gemm_gru<2>(xA,  wsl + WS_WI0, wave, lane, aR, aZ, aNi);
    gemm_gru<8>(hA0, wsl + WS_WH0, wave, lane, aR, aZ, aNh);

#pragma unroll
    for (int u = 0; u < 2; ++u) {
#pragma unroll
      for (int j = 0; j < 8; ++j) {
        float r = sigm_(aR[u][j] + vbr0[u]);
        float z = sigm_(aZ[u][j] + vbz0[u]);
        float n = tanhf(aNi[u][j] + vbin0[u] + r * (aNh[u][j] + vbhn0[u]));
        h0r[u][j] = (1.0f - z) * n + z * h0r[u][j];
      }
    }

    __syncthreads();  // all waves finished reading old hA0
    // write new h0 A-fragment (wave w owns k-tile w); C->A layout transpose
#pragma unroll
    for (int u = 0; u < 2; ++u) {
#pragma unroll
      for (int j = 0; j < 8; ++j) {
        int m  = rowOff + j;
        int lp = m + ((col16 & 8) ? 16 : 0);
        int e  = (col16 & 7) + u * 8;
        hA0[wave * 512 + lp * 16 + e] = (__bf16)h0r[u][j];
      }
    }
    __syncthreads();  // new hA0 ready

    // ---- layer 1: gi1 = h0_new@wi1, gh1 = h1_old@wh1 ----
    v8f cR[2] = {}, cZ[2] = {}, cNi[2] = {}, cNh[2] = {};
    gemm_gru<8>(hA0, wsl + WS_WI1, wave, lane, cR, cZ, cNi);
    gemm_gru<8>(hA1, wsl + WS_WH1, wave, lane, cR, cZ, cNh);

    gfp outp = og + (size_t)t * MROWS * HSZ;
#pragma unroll
    for (int u = 0; u < 2; ++u) {
#pragma unroll
      for (int j = 0; j < 8; ++j) {
        float r = sigm_(cR[u][j] + vbr1[u]);
        float z = sigm_(cZ[u][j] + vbz1[u]);
        float n = tanhf(cNi[u][j] + vbin1[u] + r * (cNh[u][j] + vbhn1[u]));
        float h = (1.0f - z) * n + z * h1r[u][j];
        h1r[u][j] = h;
        int row = R0 + rowOff + j;
        int col = wave * 32 + u * 16 + col16;
        outp[(size_t)row * HSZ + col] = h;   // output[t] = h1
      }
    }

    __syncthreads();  // all waves finished reading old hA1
#pragma unroll
    for (int u = 0; u < 2; ++u) {
#pragma unroll
      for (int j = 0; j < 8; ++j) {
        int m  = rowOff + j;
        int lp = m + ((col16 & 8) ? 16 : 0);
        int e  = (col16 & 7) + u * 8;
        hA1[wave * 512 + lp * 16 + e] = (__bf16)h1r[u][j];
      }
    }
    __syncthreads();  // hA1 ready for next step; xA safe to overwrite
  }

  // ---- final hidden states: hidden_out = stack([h0, h1]) ----
  size_t hb = (size_t)SEQ * MROWS * HSZ;
#pragma unroll
  for (int u = 0; u < 2; ++u) {
#pragma unroll
    for (int j = 0; j < 8; ++j) {
      int row = R0 + rowOff + j;
      int col = wave * 32 + u * 16 + col16;
      og[hb + (size_t)row * HSZ + col]                       = h0r[u][j];
      og[hb + (size_t)MROWS * HSZ + (size_t)row * HSZ + col] = h1r[u][j];
    }
  }
}

extern "C" void kernel_launch(void* const* d_in, const int* in_sizes, int n_in,
                              void* d_out, int out_size, void* d_ws, size_t ws_size,
                              hipStream_t stream) {
  const float* x   = (const float*)d_in[0];
  const float* wi0 = (const float*)d_in[1];
  const float* wh0 = (const float*)d_in[2];
  const float* bi0 = (const float*)d_in[3];
  const float* bh0 = (const float*)d_in[4];
  const float* wi1 = (const float*)d_in[5];
  const float* wh1 = (const float*)d_in[6];
  const float* bi1 = (const float*)d_in[7];
  const float* bh1 = (const float*)d_in[8];
  __bf16* ws = (__bf16*)d_ws;   // needs 1.25 MB of scratch
  float* out = (float*)d_out;

  gru_swizzle_weights<<<(WS_TOTAL + 255) / 256, 256, 0, stream>>>(wi0, wh0, wi1, wh1, ws);
  gru_persistent<<<MROWS / 16, 256, 0, stream>>>(x, bi0, bh0, bi1, bh1, ws, out);
}
